// LSTM_80315888435470
// MI455X (gfx1250) — matrix operand order repaired and checked by reading, not compiled, not run on hardware
//
#include <hip/hip_runtime.h>

// LSTM on MI455X (gfx1250): bf16 WMMA 16x16x32, fp32 accumulate.
// Phase 1: precompute input-half preactivations Xg[t] = Wx@x_t + b (parallel over T).
// Phase 2: 512 sequential step kernels: gates = Wh@h_{t-1} + Xg[t], cell update.

#define T_STEPS 512
#define I_DIM   1536
#define H_DIM   1536
#define B_DIM   64
#define K_TOT   (H_DIM + I_DIM)   // 3072

typedef __attribute__((ext_vector_type(16))) __bf16 v16bf;
typedef __attribute__((ext_vector_type(8)))  __bf16 v8bf;
typedef __attribute__((ext_vector_type(8)))  float  v8f;

// ---- WMMA fragment loaders (wave32 layouts per CDNA5 ISA 7.12.2) ----------
// A 16x32 bf16, row-major source, lane l: m=l&15, kb = (l<16)?0:8
//   e[0..7]  = A[m][kb .. kb+7]      e[8..15] = A[m][16+kb .. 16+kb+7]
__device__ __forceinline__ v16bf load_a_frag(const __bf16* __restrict__ A,
                                             int lda, int lane) {
  const int m  = lane & 15;
  const int kb = (lane & 16) ? 8 : 0;
  const __bf16* p = A + (size_t)m * lda + kb;
  v8bf lo = *(const v8bf*)(p);
  v8bf hi = *(const v8bf*)(p + 16);
  return __builtin_shufflevector(lo, hi, 0,1,2,3,4,5,6,7,8,9,10,11,12,13,14,15);
}

// B 32x16 bf16 from an [N][K] (transposed) source, lane l: n=l&15, ko=(l<16)?0:8
//   e[0..7] = src[n][ko .. ko+7]      e[8..15] = src[n][16+ko .. 16+ko+7]
__device__ __forceinline__ v16bf load_b_frag(const __bf16* __restrict__ Bm,
                                             int ldb, int lane) {
  const int n  = lane & 15;
  const int ko = (lane & 16) ? 8 : 0;
  const __bf16* p = Bm + (size_t)n * ldb + ko;
  v8bf lo = *(const v8bf*)(p);
  v8bf hi = *(const v8bf*)(p + 16);
  return __builtin_shufflevector(lo, hi, 0,1,2,3,4,5,6,7,8,9,10,11,12,13,14,15);
}

__device__ __forceinline__ float sigmoid_f(float x) {
  return 1.0f / (1.0f + __expf(-x));
}

// ---- Conversion kernels ----------------------------------------------------
__global__ void lstm_convW(const float* __restrict__ Wf, const float* __restrict__ Wi,
                           const float* __restrict__ Wo, const float* __restrict__ Wc,
                           __bf16* __restrict__ Wb) {
  size_t idx = (size_t)blockIdx.x * blockDim.x + threadIdx.x;
  const size_t per = (size_t)H_DIM * K_TOT;
  if (idx >= per) return;
  Wb[0 * per + idx] = (__bf16)Wf[idx];
  Wb[1 * per + idx] = (__bf16)Wi[idx];
  Wb[2 * per + idx] = (__bf16)Wo[idx];
  Wb[3 * per + idx] = (__bf16)Wc[idx];
}

// x [T][I][B] fp32 -> xbT [T][B][I] bf16  (K-contiguous per batch column)
__global__ void lstm_convX(const float* __restrict__ x, __bf16* __restrict__ xbT) {
  size_t idx = (size_t)blockIdx.x * blockDim.x + threadIdx.x;
  const size_t total = (size_t)T_STEPS * I_DIM * B_DIM;
  if (idx >= total) return;
  size_t b = idx % B_DIM;
  size_t i = (idx / B_DIM) % I_DIM;
  size_t t = idx / ((size_t)B_DIM * I_DIM);
  float v = __builtin_nontemporal_load(&x[idx]);
  xbT[(t * B_DIM + b) * I_DIM + i] = (__bf16)v;
}

__global__ void lstm_init(const float* __restrict__ h0, const float* __restrict__ c0,
                          __bf16* __restrict__ hbT, float* __restrict__ c) {
  int idx = blockIdx.x * blockDim.x + threadIdx.x;
  if (idx >= H_DIM * B_DIM) return;
  int n = idx % B_DIM;
  int m = idx / B_DIM;
  c[idx] = c0[idx];
  hbT[(size_t)n * H_DIM + m] = (__bf16)h0[idx];
}

// ---- Phase 1: Xg[t][g*H+m][n] = Wx_g @ x_t + b_g  (parallel over T) --------
// One wave per 16x16 tile; 4 gate accumulators share one B fragment.
__global__ void lstm_xgemm(const __bf16* __restrict__ Wb, const __bf16* __restrict__ xbT,
                           const float* __restrict__ bF, const float* __restrict__ bI,
                           const float* __restrict__ bO, const float* __restrict__ bC,
                           float* __restrict__ Xg) {
  const int m0   = blockIdx.x * 16;        // 0..H-16
  const int n0   = blockIdx.y * 16;        // 0..B-16
  const int t    = blockIdx.z;
  const int lane = threadIdx.x;

  v8f acc0 = {}, acc1 = {}, acc2 = {}, acc3 = {};
  const __bf16* Bbase = xbT + ((size_t)t * B_DIM + n0) * I_DIM;
  const size_t  per   = (size_t)H_DIM * K_TOT;
  const __bf16* A0 = Wb + 0 * per + (size_t)m0 * K_TOT + H_DIM;  // input half cols
  const __bf16* A1 = Wb + 1 * per + (size_t)m0 * K_TOT + H_DIM;
  const __bf16* A2 = Wb + 2 * per + (size_t)m0 * K_TOT + H_DIM;
  const __bf16* A3 = Wb + 3 * per + (size_t)m0 * K_TOT + H_DIM;

  for (int k = 0; k < I_DIM; k += 32) {
    v16bf bf = load_b_frag(Bbase + k, I_DIM, lane);
    v16bf a;
    a = load_a_frag(A0 + k, K_TOT, lane);
    acc0 = __builtin_amdgcn_wmma_f32_16x16x32_bf16(false, a, false, bf, (short)0, acc0, false, false);
    a = load_a_frag(A1 + k, K_TOT, lane);
    acc1 = __builtin_amdgcn_wmma_f32_16x16x32_bf16(false, a, false, bf, (short)0, acc1, false, false);
    a = load_a_frag(A2 + k, K_TOT, lane);
    acc2 = __builtin_amdgcn_wmma_f32_16x16x32_bf16(false, a, false, bf, (short)0, acc2, false, false);
    a = load_a_frag(A3 + k, K_TOT, lane);
    acc3 = __builtin_amdgcn_wmma_f32_16x16x32_bf16(false, a, false, bf, (short)0, acc3, false, false);
  }

  const int n  = lane & 15;
  const int mb = (lane >> 4) * 8;
  float* base = Xg + (size_t)t * 4 * H_DIM * B_DIM;
#pragma unroll
  for (int v = 0; v < 8; ++v) {
    int m = m0 + mb + v;
    size_t idx = (size_t)m * B_DIM + n0 + n;
    // Non-temporal: Xg (805 MB) is streamed once; keep weights resident in L2.
    __builtin_nontemporal_store(acc0[v] + bF[m], &base[0 * (size_t)H_DIM * B_DIM + idx]);
    __builtin_nontemporal_store(acc1[v] + bI[m], &base[1 * (size_t)H_DIM * B_DIM + idx]);
    __builtin_nontemporal_store(acc2[v] + bO[m], &base[2 * (size_t)H_DIM * B_DIM + idx]);
    __builtin_nontemporal_store(acc3[v] + bC[m], &base[3 * (size_t)H_DIM * B_DIM + idx]);
  }
}

// ---- Phase 2: one step. gates = Wh@h_{t-1} + Xg[t]; cell update in-register.
__global__ void lstm_step(const __bf16* __restrict__ Wb, const __bf16* __restrict__ hbT_in,
                          __bf16* __restrict__ hbT_out, const float* __restrict__ Xg_t,
                          float* __restrict__ c, float* __restrict__ out_t) {
  const int m0   = blockIdx.x * 16;
  const int n0   = blockIdx.y * 16;
  const int lane = threadIdx.x;

  v8f acc0 = {}, acc1 = {}, acc2 = {}, acc3 = {};
  const __bf16* Bbase = hbT_in + (size_t)n0 * H_DIM;
  const size_t  per   = (size_t)H_DIM * K_TOT;
  const __bf16* A0 = Wb + 0 * per + (size_t)m0 * K_TOT;   // recurrent half cols [0,H)
  const __bf16* A1 = Wb + 1 * per + (size_t)m0 * K_TOT;
  const __bf16* A2 = Wb + 2 * per + (size_t)m0 * K_TOT;
  const __bf16* A3 = Wb + 3 * per + (size_t)m0 * K_TOT;

  for (int k = 0; k < H_DIM; k += 32) {
    v16bf bf = load_b_frag(Bbase + k, H_DIM, lane);
    v16bf a;
    a = load_a_frag(A0 + k, K_TOT, lane);
    acc0 = __builtin_amdgcn_wmma_f32_16x16x32_bf16(false, a, false, bf, (short)0, acc0, false, false);
    a = load_a_frag(A1 + k, K_TOT, lane);
    acc1 = __builtin_amdgcn_wmma_f32_16x16x32_bf16(false, a, false, bf, (short)0, acc1, false, false);
    a = load_a_frag(A2 + k, K_TOT, lane);
    acc2 = __builtin_amdgcn_wmma_f32_16x16x32_bf16(false, a, false, bf, (short)0, acc2, false, false);
    a = load_a_frag(A3 + k, K_TOT, lane);
    acc3 = __builtin_amdgcn_wmma_f32_16x16x32_bf16(false, a, false, bf, (short)0, acc3, false, false);
  }

  const int n  = lane & 15;
  const int mb = (lane >> 4) * 8;
  const size_t plane = (size_t)H_DIM * B_DIM;
#pragma unroll
  for (int v = 0; v < 8; ++v) {
    int m = m0 + mb + v;
    size_t idx = (size_t)m * B_DIM + n0 + n;
    float pf = acc0[v] + __builtin_nontemporal_load(&Xg_t[0 * plane + idx]);
    float pi = acc1[v] + __builtin_nontemporal_load(&Xg_t[1 * plane + idx]);
    float po = acc2[v] + __builtin_nontemporal_load(&Xg_t[2 * plane + idx]);
    float pc = acc3[v] + __builtin_nontemporal_load(&Xg_t[3 * plane + idx]);
    float f  = sigmoid_f(pf);
    float ig = sigmoid_f(pi);
    float o  = sigmoid_f(po);
    float cc = tanhf(pc);
    float cn = f * c[idx] + ig * cc;
    float hn = o * tanhf(cn);
    c[idx] = cn;                                      // c stays hot in L2
    __builtin_nontemporal_store(hn, &out_t[idx]);     // hs never re-read
    hbT_out[(size_t)(n0 + n) * H_DIM + m] = (__bf16)hn;  // bf16, [n][k] for next step
  }
}

// ---- Host-side orchestration ----------------------------------------------
extern "C" void kernel_launch(void* const* d_in, const int* in_sizes, int n_in,
                              void* d_out, int out_size, void* d_ws, size_t ws_size,
                              hipStream_t stream) {
  const float* x   = (const float*)d_in[0];
  const float* Wf  = (const float*)d_in[1];
  const float* bF  = (const float*)d_in[2];
  const float* Wi  = (const float*)d_in[3];
  const float* bI  = (const float*)d_in[4];
  const float* Wo  = (const float*)d_in[5];
  const float* bO  = (const float*)d_in[6];
  const float* Wc  = (const float*)d_in[7];
  const float* bC  = (const float*)d_in[8];
  const float* h0  = (const float*)d_in[9];
  const float* c0  = (const float*)d_in[10];
  float* out = (float*)d_out;

  char* ws = (char*)d_ws;
  size_t off = 0;
  __bf16* Wb  = (__bf16*)(ws + off); off += (size_t)4 * H_DIM * K_TOT * sizeof(__bf16);   // 37.7 MB
  __bf16* xbT = (__bf16*)(ws + off); off += (size_t)T_STEPS * B_DIM * I_DIM * sizeof(__bf16); // 100 MB
  float*  Xg  = (float*)(ws + off);  off += (size_t)T_STEPS * 4 * H_DIM * B_DIM * sizeof(float); // 805 MB
  __bf16* hA  = (__bf16*)(ws + off); off += (size_t)B_DIM * H_DIM * sizeof(__bf16);
  __bf16* hB  = (__bf16*)(ws + off); off += (size_t)B_DIM * H_DIM * sizeof(__bf16);
  float*  cws = (float*)(ws + off);  off += (size_t)H_DIM * B_DIM * sizeof(float);
  (void)ws_size; (void)in_sizes; (void)n_in; (void)out_size;

  {
    size_t n = (size_t)H_DIM * K_TOT;
    lstm_convW<<<dim3((unsigned)((n + 255) / 256)), 256, 0, stream>>>(Wf, Wi, Wo, Wc, Wb);
  }
  {
    size_t n = (size_t)T_STEPS * I_DIM * B_DIM;
    lstm_convX<<<dim3((unsigned)((n + 255) / 256)), 256, 0, stream>>>(x, xbT);
  }
  {
    size_t n = (size_t)H_DIM * B_DIM;
    lstm_init<<<dim3((unsigned)((n + 255) / 256)), 256, 0, stream>>>(h0, c0, hA, cws);
  }

  // Phase 1: all input-half GEMMs at once (parallel over T).
  lstm_xgemm<<<dim3(H_DIM / 16, B_DIM / 16, T_STEPS), 32, 0, stream>>>(
      Wb, xbT, bF, bI, bO, bC, Xg);

  // Phase 2: 512 dependent steps; h ping-pongs between two bf16 transposed buffers.
  for (int t = 0; t < T_STEPS; ++t) {
    const __bf16* hin = (t & 1) ? hB : hA;
    __bf16*       hout = (t & 1) ? hA : hB;
    lstm_step<<<dim3(H_DIM / 16, B_DIM / 16), 32, 0, stream>>>(
        Wb, hin, hout, Xg + (size_t)t * 4 * H_DIM * B_DIM, cws,
        out + (size_t)t * H_DIM * B_DIM);
  }
}
